// DilConv_81466939671248
// MI455X (gfx1250) — matrix operand order repaired and checked by reading, not compiled, not run on hardware
//
#include <hip/hip_runtime.h>
#include <hip/hip_bf16.h>

// ---------------------------------------------------------------------------
// Problem constants (reference: N,H,W,C = 32,56,56,256; K=3, RATE=2, STRIDE=1)
// ---------------------------------------------------------------------------
#define BN   32
#define BH   56
#define BW   56
#define BC   256
#define RATE 2
#define M_TOTAL (BN * BH * BW)          // 100352 output pixels
#define BN_EPS 1e-3f

// GEMM tiling: block = 256 threads = 8 waves (wave32).
// Block tile: 32 (M) x 256 (N = full) -> A tile read from HBM exactly once.
// Wave grid 2x4: wave (mw,nq) owns a 16(M) x 64(N) strip = 4 accumulators.
// K-loop = 256/32 = 8 chunks x 4 WMMAs = 32 v_wmma_f32_16x16x32_bf16 / wave.
#define MT 32

typedef __attribute__((ext_vector_type(16))) __bf16 v16bf;
typedef __attribute__((ext_vector_type(8)))  __bf16 v8bf;
typedef __attribute__((ext_vector_type(8)))  float  v8f;
typedef __attribute__((ext_vector_type(4)))  unsigned int u32x4;
typedef __attribute__((ext_vector_type(8)))  unsigned int u32x8;

// ---------------------------------------------------------------------------
// fp32 -> bf16 (round to nearest even, ignoring NaN edge case)
// ---------------------------------------------------------------------------
__device__ __forceinline__ unsigned short f2bf(float f) {
  unsigned int u = __builtin_bit_cast(unsigned int, f);
  u += 0x7FFFu + ((u >> 16) & 1u);
  return (unsigned short)(u >> 16);
}

// ---------------------------------------------------------------------------
// Kernel 1: transpose + convert pointwise weights: [K][N] f32 -> [N][K] bf16.
// ---------------------------------------------------------------------------
__global__ void prep_weights(const float* __restrict__ pwk,
                             unsigned short* __restrict__ Bt) {
  int n = blockIdx.x;
  int k = threadIdx.x;
  Bt[n * BC + k] = f2bf(pwk[k * BC + n]);
}

// ---------------------------------------------------------------------------
// Kernel 2: fold pw_bias + BN: out = conv * scale[c] + shift[c]
// ---------------------------------------------------------------------------
__global__ void prep_bn(const float* __restrict__ gamma,
                        const float* __restrict__ beta,
                        const float* __restrict__ mean,
                        const float* __restrict__ var,
                        const float* __restrict__ pwb,
                        float* __restrict__ scale,
                        float* __restrict__ shift) {
  int c = threadIdx.x;
  float sc = gamma[c] * rsqrtf(var[c] + BN_EPS);
  scale[c] = sc;
  shift[c] = (pwb[c] - mean[c]) * sc + beta[c];
}

// ---------------------------------------------------------------------------
// Kernel 3: fused ReLU + dilated 3x3 depthwise conv + bias -> bf16 A matrix.
//   A is [M_TOTAL][256] bf16 row-major. 1 thread = 1 pixel x 4 channels.
// ---------------------------------------------------------------------------
__global__ void dw_relu_conv(const float* __restrict__ x,
                             const float* __restrict__ dwk,
                             const float* __restrict__ dwb,
                             unsigned short* __restrict__ A) {
  const int tid = threadIdx.x;
  const int pix = blockIdx.x * 4 + (tid >> 6);   // global pixel index m
  const int c   = (tid & 63) * 4;                // channel group

  const int n   = pix / (BH * BW);
  const int rem = pix - n * (BH * BW);
  const int h   = rem / BW;
  const int w   = rem - h * BW;

  float4 acc = *(const float4*)(dwb + c);

#pragma unroll
  for (int kh = 0; kh < 3; ++kh) {
#pragma unroll
    for (int kw = 0; kw < 3; ++kw) {
      const int ih = h + (kh - 1) * RATE;
      const int iw = w + (kw - 1) * RATE;
      if (ih >= 0 && ih < BH && iw >= 0 && iw < BW) {
        const size_t xi = ((size_t)((n * BH + ih) * BW + iw)) * BC + c;
        float4 xv = *(const float4*)(x + xi);
        xv.x = fmaxf(xv.x, 0.f); xv.y = fmaxf(xv.y, 0.f);
        xv.z = fmaxf(xv.z, 0.f); xv.w = fmaxf(xv.w, 0.f);
        const float4 kv = *(const float4*)(dwk + (kh * 3 + kw) * BC + c);
        acc.x = fmaf(xv.x, kv.x, acc.x);
        acc.y = fmaf(xv.y, kv.y, acc.y);
        acc.z = fmaf(xv.z, kv.z, acc.z);
        acc.w = fmaf(xv.w, kv.w, acc.w);
      }
    }
  }

  ushort4 o;
  o.x = f2bf(acc.x); o.y = f2bf(acc.y); o.z = f2bf(acc.z); o.w = f2bf(acc.w);
  *(ushort4*)(A + (size_t)pix * BC + c) = o;
}

// ---------------------------------------------------------------------------
// Kernel 4: pointwise GEMM via WMMA bf16 -> f32, fused BN epilogue.
//   out[m][n] = ( sum_k A[m][k] * Bt[n][k] ) * scale[n] + shift[n]
// grid = M_TOTAL/32 blocks, 256 threads (8 waves).
// A tile (32 x 256 bf16 = 16 KB) DMA'd into LDS by the Tensor Data Mover
// (tensor_load_to_lds, TENSORcnt); B served from L2 (128 KB, resident).
// ---------------------------------------------------------------------------
__global__ void __launch_bounds__(256)
pw_gemm_wmma(const unsigned short* __restrict__ A,
             const unsigned short* __restrict__ Bt,
             const float* __restrict__ scale,
             const float* __restrict__ shift,
             float* __restrict__ out) {
  __shared__ __align__(16) unsigned short ldsA[MT * BC];   // 16 KB

  const int m0  = blockIdx.x * MT;
  const int tid = threadIdx.x;
  const int wave = tid >> 5;
  const int lane = tid & 31;

  // --- TDM: async DMA of the 32x256 bf16 A tile into LDS -------------------
  if (wave == 0) {
    const unsigned long long ga =
        (unsigned long long)(const void*)(A + (size_t)m0 * BC);
    const unsigned lds_addr = (unsigned)(size_t)(void*)ldsA;

    // D# group 0 (128b): count=1 | lds_addr | global_addr[56:0] | type=2
    u32x4 g0;
    g0[0] = 1u;                                            // count = 1 (valid)
    g0[1] = lds_addr;                                      // LDS byte address
    g0[2] = (unsigned)ga;                                  // global_addr[31:0]
    g0[3] = (unsigned)((ga >> 32) & 0x01FFFFFFu) | (2u << 30); // [56:32]|type=2

    // D# group 1 (256b): data_size=2B, tensor 256 x M_TOTAL, tile 256 x 32,
    // dim0 stride = 256 elements (row-major, K contiguous).
    u32x8 g1;
    g1[0] = 1u << 16;                                      // data_size=1 (2B)
    g1[1] = ((unsigned)BC & 0xFFFFu) << 16;                // tensor_dim0 lo16
    g1[2] = (((unsigned)BC >> 16) & 0xFFFFu)               // tensor_dim0 hi16
          | (((unsigned)M_TOTAL & 0xFFFFu) << 16);         // tensor_dim1 lo16
    g1[3] = (((unsigned)M_TOTAL >> 16) & 0xFFFFu)          // tensor_dim1 hi16
          | (((unsigned)BC & 0xFFFFu) << 16);              // tile_dim0 = 256
    g1[4] = (unsigned)MT;                                  // tile_dim1 = 32
    g1[5] = (unsigned)BC;                                  // dim0_stride lo32
    g1[6] = 0u;                                            // stride hi / dim1_stride lo
    g1[7] = 0u;

    asm volatile("tensor_load_to_lds %0, %1" :: "s"(g0), "s"(g1) : "memory");
    __builtin_amdgcn_s_wait_tensorcnt(0);
  }
  __syncthreads();

  // --- wave / lane geometry ------------------------------------------------
  const int mw   = wave >> 2;          // 0..1 -> M sub-tile
  const int nq   = wave & 3;           // 0..3 -> N quadrant (64 cols)
  const int half = lane >> 4;          // 0 = lanes 0-15, 1 = lanes 16-31
  const int lr   = lane & 15;

  // A fragment (16-bit A 16x32 layout):
  //   lanes 0-15 : row M=lr, K = kc+[0..7] and kc+[16..23]
  //   lanes 16-31: row M=lr, K = kc+[8..15] and kc+[24..31]
  const unsigned short* aRow = ldsA + (mw * 16 + lr) * BC + half * 8;

  // B fragments (16-bit B 32x16 layout), 4 tiles of 16 N each:
  //   lanes 0-15 : col N, K = kc+[0..7] and kc+[16..23] ; lanes 16-31: K = kc+[8..15] and kc+[24..31]
  const unsigned short* bBase = Bt + (size_t)(nq * 64 + lr) * BC + half * 8;

  v8f acc0 = {}, acc1 = {}, acc2 = {}, acc3 = {};

#pragma unroll
  for (int kc = 0; kc < BC; kc += 32) {
    v8bf a_lo = *(const v8bf*)(aRow + kc);        // ds_load_b128
    v8bf a_hi = *(const v8bf*)(aRow + kc + 16);
    v16bf a = __builtin_shufflevector(a_lo, a_hi,
        0, 1, 2, 3, 4, 5, 6, 7, 8, 9, 10, 11, 12, 13, 14, 15);

    v16bf b[4];
#pragma unroll
    for (int j = 0; j < 4; ++j) {
      const unsigned short* bp = bBase + (size_t)j * 16 * BC + kc;
      v8bf b_lo = *(const v8bf*)(bp);             // global_load_b128 (L2-hot)
      v8bf b_hi = *(const v8bf*)(bp + 16);
      b[j] = __builtin_shufflevector(b_lo, b_hi,
          0, 1, 2, 3, 4, 5, 6, 7, 8, 9, 10, 11, 12, 13, 14, 15);
    }

    acc0 = __builtin_amdgcn_wmma_f32_16x16x32_bf16(false, a, false, b[0],
                                                   (short)0, acc0, false, false);
    acc1 = __builtin_amdgcn_wmma_f32_16x16x32_bf16(false, a, false, b[1],
                                                   (short)0, acc1, false, false);
    acc2 = __builtin_amdgcn_wmma_f32_16x16x32_bf16(false, a, false, b[2],
                                                   (short)0, acc2, false, false);
    acc3 = __builtin_amdgcn_wmma_f32_16x16x32_bf16(false, a, false, b[3],
                                                   (short)0, acc3, false, false);
  }

  // --- epilogue: C/D layout -> scale/shift -> fp32 store -------------------
  //   VGPR r: lanes 0-15 hold (M = base+r, N = ncol); lanes 16-31 (M = base+8+r)
  const int m_base = m0 + mw * 16 + half * 8;
  v8f accs[4] = {acc0, acc1, acc2, acc3};

#pragma unroll
  for (int j = 0; j < 4; ++j) {
    const int ncol = nq * 64 + j * 16 + lr;
    const float sc = scale[ncol];
    const float sh = shift[ncol];
#pragma unroll
    for (int r = 0; r < 8; ++r) {
      out[(size_t)(m_base + r) * BC + ncol] = fmaf(accs[j][r], sc, sh);
    }
  }
}

// ---------------------------------------------------------------------------
// Launch
// ---------------------------------------------------------------------------
extern "C" void kernel_launch(void* const* d_in, const int* in_sizes, int n_in,
                              void* d_out, int out_size, void* d_ws, size_t ws_size,
                              hipStream_t stream) {
  const float* x     = (const float*)d_in[0];
  const float* dwk   = (const float*)d_in[1];
  const float* dwb   = (const float*)d_in[2];
  const float* pwk   = (const float*)d_in[3];
  const float* pwb   = (const float*)d_in[4];
  const float* gamma = (const float*)d_in[5];
  const float* beta  = (const float*)d_in[6];
  const float* mean  = (const float*)d_in[7];
  const float* var   = (const float*)d_in[8];
  float* out = (float*)d_out;

  // workspace layout
  const size_t A_BYTES  = (size_t)M_TOTAL * BC * sizeof(unsigned short); // 51.4 MB
  const size_t BT_BYTES = (size_t)BC * BC * sizeof(unsigned short);      // 128 KB
  char* ws = (char*)d_ws;
  unsigned short* A     = (unsigned short*)ws;
  unsigned short* Btw   = (unsigned short*)(ws + A_BYTES);
  float*          scale = (float*)(ws + A_BYTES + BT_BYTES);
  float*          shift = scale + BC;

  prep_weights<<<BC, BC, 0, stream>>>(pwk, Btw);
  prep_bn<<<1, BC, 0, stream>>>(gamma, beta, mean, var, pwb, scale, shift);
  dw_relu_conv<<<M_TOTAL / 4, 256, 0, stream>>>(x, dwk, dwb, A);

  pw_gemm_wmma<<<M_TOTAL / MT, 256, 0, stream>>>(A, Btw, scale, shift, out);
}
